// GMCF_42004780155451
// MI455X (gfx1250) — matrix operand order repaired and checked by reading, not compiled, not run on hardware
//
#include <hip/hip_runtime.h>
#include <hip/hip_bf16.h>
#include <math.h>

typedef __attribute__((ext_vector_type(16))) __bf16 v16bf;
typedef __attribute__((ext_vector_type(8)))  float  v8f;

#define NT    40960     // total nodes = B*N
#define GDIM  256
#define GHID  1024
#define NB    4096      // batch (graphs)
#define NEI   172032    // inner edges
#define NEO   196608    // outer edges

__device__ __forceinline__ v8f bcast8(float b) {
  v8f c;
#pragma unroll
  for (int i = 0; i < 8; ++i) c[i] = b;
  return c;
}

__device__ __forceinline__ float sigm(float x) { return 1.f / (1.f + __expf(-x)); }

// B-fragment (KxN = 32x16 bf16): lane -> column lane%16; elems 0..7 -> K = 8*half+j, elems 8..15 -> K = 16+8*half+j.
// For act @ W^T with W row-major (rows = output cols, K contiguous) this is
// two 16-byte contiguous runs per lane.
__device__ __forceinline__ v16bf load_b_frag(const __bf16* __restrict__ W, int row0, int K,
                                             int kblock, int lane) {
  int col = lane & 15, half = lane >> 4;
  const __bf16* p = W + (size_t)(row0 + col) * K + kblock * 32 + half * 8;
  v16bf r;
#pragma unroll
  for (int j = 0; j < 8; ++j) { r[j] = p[j]; r[j + 8] = p[16 + j]; }
  return r;
}

__device__ __forceinline__ v8f wmma_bf16(v16bf a, v16bf b, v8f c) {
  return __builtin_amdgcn_wmma_f32_16x16x32_bf16(false, a, false, b, (short)0, c, false, false);
}

// ---------------------------------------------------------------- utilities
__global__ void zero_kernel(float* __restrict__ p, long n) {
  long i = (long)blockIdx.x * blockDim.x + threadIdx.x;
  if (i < n) p[i] = 0.f;
}

__global__ void cvt_kernel(const float* __restrict__ s, __bf16* __restrict__ d, int n) {
  int i = blockIdx.x * blockDim.x + threadIdx.x;
  if (i < n) d[i] = (__bf16)s[i];
}

__global__ void gather_kernel(const float4* __restrict__ emb, const int* __restrict__ nid,
                              float4* __restrict__ x) {
  long i = (long)blockIdx.x * blockDim.x + threadIdx.x;   // NT*64 threads
  long node = i >> 6; int q = (int)(i & 63);
  x[(node << 6) + q] = emb[((size_t)nid[node] << 6) + q];
}

__global__ void copy4_kernel(const float4* __restrict__ s, float4* __restrict__ d, long n) {
  long i = (long)blockIdx.x * blockDim.x + threadIdx.x;
  if (i < n) d[i] = s[i];
}

__global__ void nodew_kernel(const int* __restrict__ nid, const int* __restrict__ batch,
                             const float* __restrict__ nw, float* __restrict__ sw) {
  int i = blockIdx.x * blockDim.x + threadIdx.x;
  if (i < NT) atomicAdd(sw + batch[i], nw[nid[i]]);
}

// outer products: outer_sum[do] += x[do]*x[so]  (8 dims per thread)
__global__ void outer_kernel(const float* __restrict__ x, const int* __restrict__ eso,
                             const int* __restrict__ edo, float* __restrict__ outerS) {
  long i = (long)blockIdx.x * blockDim.x + threadIdx.x;   // NEO*32 threads
  long e = i >> 5; int part = (int)(i & 31);
  int so_ = eso[e], do_ = edo[e];
  const float* rs = x + (size_t)so_ * GDIM + part * 8;
  const float* rd = x + (size_t)do_ * GDIM + part * 8;
  float* out = outerS + (size_t)do_ * GDIM + part * 8;
#pragma unroll
  for (int j = 0; j < 8; ++j) atomicAdd(out + j, rs[j] * rd[j]);
}

// ------------------------------------------------------------ edge MLP (WMMA)
// One wave = 16 edges. pw = x[dst]*x[src]; h1 = relu(pw@W1^T+b1); msg = h1@W2^T+b2
// msg atomically accumulated into inner segment sums.
__global__ void __launch_bounds__(256, 1)
edge_mlp_kernel(const float* __restrict__ x, const int* __restrict__ esrc,
                const int* __restrict__ edst,
                const __bf16* __restrict__ w1, const float* __restrict__ b1,
                const __bf16* __restrict__ w2, const float* __restrict__ b2,
                float* __restrict__ innerS) {
  __shared__ __bf16 lds[8 * 16 * 32];
  int lane = threadIdx.x & 31;
  int wid  = threadIdx.x >> 5;
  __bf16* h1t = lds + wid * 512;
  int tile  = blockIdx.x * 8 + wid;
  int ebase = tile * 16;
  int m = lane & 15, half = lane >> 4, n = m;

  __builtin_prefetch(w1, 0, 3);
  __builtin_prefetch(w2, 0, 3);

  // A (pw) fragments: 16x256 across 8 K-blocks; ISA 16-bit A layout:
  // elems 0..7 -> K = 8*half+j ; elems 8..15 -> K = 16+8*half+j  (within 32-block)
  int er = ebase + m;
  const float* rs = x + (size_t)esrc[er] * GDIM;
  const float* rd = x + (size_t)edst[er] * GDIM;
  v16bf apw[8];
#pragma unroll
  for (int k = 0; k < 8; ++k) {
    int b0 = k * 32 + half * 8;
#pragma unroll
    for (int j = 0; j < 8; ++j) {
      apw[k][j]     = (__bf16)(rs[b0 + j] * rd[b0 + j]);
      apw[k][j + 8] = (__bf16)(rs[b0 + 16 + j] * rd[b0 + 16 + j]);
    }
  }

  int drow[8];
#pragma unroll
  for (int r = 0; r < 8; ++r) drow[r] = edst[ebase + r + 8 * half];

  // msg accumulator 16x256, bias-initialized
  v8f macc[16];
#pragma unroll
  for (int t = 0; t < 16; ++t) macc[t] = bcast8(b2[t * 16 + n]);

  for (int hc = 0; hc < 32; ++hc) {          // 32 HID-chunks of 32 columns
    v8f c0 = bcast8(b1[hc * 32 + n]);
    v8f c1 = bcast8(b1[hc * 32 + 16 + n]);
#pragma unroll
    for (int k = 0; k < 8; ++k) {
      c0 = wmma_bf16(apw[k], load_b_frag(w1, hc * 32,      GDIM, k, lane), c0);
      c1 = wmma_bf16(apw[k], load_b_frag(w1, hc * 32 + 16, GDIM, k, lane), c1);
    }
    // relu + cross-lane transpose (C layout -> A layout) through per-wave LDS
#pragma unroll
    for (int r = 0; r < 8; ++r) {
      int rowm = r + 8 * half;
      float v0 = c0[r]; v0 = v0 > 0.f ? v0 : 0.f;
      float v1 = c1[r]; v1 = v1 > 0.f ? v1 : 0.f;
      h1t[rowm * 32 + n]      = (__bf16)v0;
      h1t[rowm * 32 + 16 + n] = (__bf16)v1;
    }
    __builtin_amdgcn_wave_barrier();   // keep DS store->load program order (same wave, in-order LDS)
    v16bf ah1;
    {
      const __bf16* pr = h1t + m * 32;
#pragma unroll
      for (int j = 0; j < 8; ++j) {
        ah1[j]     = pr[8 * half + j];
        ah1[j + 8] = pr[16 + 8 * half + j];
      }
    }
#pragma unroll
    for (int t = 0; t < 16; ++t)
      macc[t] = wmma_bf16(ah1, load_b_frag(w2, t * 16, GHID, hc, lane), macc[t]);
  }

#pragma unroll
  for (int t = 0; t < 16; ++t)
#pragma unroll
    for (int r = 0; r < 8; ++r)
      atomicAdd(innerS + (size_t)drow[r] * GDIM + t * 16 + n, macc[t][r]);
}

// ------------------------------------------------------------- GRU step (WMMA)
// mode 0: xt raw (x); mode 1: xt = inner sums, scale 1/3|1/5; mode 2: outer sums, 1/6|1/4
__global__ void __launch_bounds__(256, 1)
gru_kernel(const float* __restrict__ xt, float* __restrict__ h,
           const __bf16* __restrict__ wih, const __bf16* __restrict__ whh,
           const float* __restrict__ bih, const float* __restrict__ bhh, int mode) {
  int lane = threadIdx.x & 31;
  int wid  = threadIdx.x >> 5;
  int tile = blockIdx.x * 8 + wid;
  int base = tile * 16;
  int m = lane & 15, half = lane >> 4, n = m;
  int node = base + m;
  int pos  = node % 10;
  float scale = 1.f;
  if (mode == 1)      scale = (pos < 4) ? (1.f / 3.f) : (1.f / 5.f);
  else if (mode == 2) scale = (pos < 4) ? (1.f / 6.f) : (1.f / 4.f);

  const float* rx = xt + (size_t)node * GDIM;
  const float* rh = h  + (size_t)node * GDIM;
  v16bf ax[8], ah[8];
#pragma unroll
  for (int k = 0; k < 8; ++k) {
    int b0 = k * 32 + half * 8;
#pragma unroll
    for (int j = 0; j < 8; ++j) {
      ax[k][j]     = (__bf16)(rx[b0 + j] * scale);
      ax[k][j + 8] = (__bf16)(rx[b0 + 16 + j] * scale);
      ah[k][j]     = (__bf16)rh[b0 + j];
      ah[k][j + 8] = (__bf16)rh[b0 + 16 + j];
    }
  }

  for (int t = 0; t < 16; ++t) {
    int col = t * 16 + n;
    v8f cir  = bcast8(bih[col]);
    v8f ciz  = bcast8(bih[256 + col]);
    v8f cin  = bcast8(bih[512 + col]);
    v8f chr_ = bcast8(bhh[col]);
    v8f chz  = bcast8(bhh[256 + col]);
    v8f chn  = bcast8(bhh[512 + col]);
#pragma unroll
    for (int k = 0; k < 8; ++k) {
      cir  = wmma_bf16(ax[k], load_b_frag(wih, t * 16,       GDIM, k, lane), cir);
      ciz  = wmma_bf16(ax[k], load_b_frag(wih, 256 + t * 16, GDIM, k, lane), ciz);
      cin  = wmma_bf16(ax[k], load_b_frag(wih, 512 + t * 16, GDIM, k, lane), cin);
      chr_ = wmma_bf16(ah[k], load_b_frag(whh, t * 16,       GDIM, k, lane), chr_);
      chz  = wmma_bf16(ah[k], load_b_frag(whh, 256 + t * 16, GDIM, k, lane), chz);
      chn  = wmma_bf16(ah[k], load_b_frag(whh, 512 + t * 16, GDIM, k, lane), chn);
    }
#pragma unroll
    for (int r = 0; r < 8; ++r) {
      float rg = sigm(cir[r] + chr_[r]);
      float zg = sigm(ciz[r] + chz[r]);
      float ng = tanhf(cin[r] + rg * chn[r]);
      size_t off = (size_t)(base + r + 8 * half) * GDIM + col;
      float hv = h[off];
      h[off] = (1.f - zg) * ng + zg * hv;   // wave owns these rows -> race-free
    }
  }
}

// --------------------------------------------------- final per-graph readout
// wave per graph: user mean (4 nodes) . item mean (6 nodes) + sum_weight -> sigmoid
__global__ void __launch_bounds__(256, 1)
final_kernel(const float* __restrict__ h, const float* __restrict__ sw,
             float* __restrict__ y) {
  int lane = threadIdx.x & 31;
  int wid  = threadIdx.x >> 5;
  int g = blockIdx.x * 8 + wid;
  float us[8], is_[8];
#pragma unroll
  for (int j = 0; j < 8; ++j) { us[j] = 0.f; is_[j] = 0.f; }
#pragma unroll
  for (int p = 0; p < 10; ++p) {
    const float* hr = h + ((size_t)g * 10 + p) * GDIM + lane * 8;
    if (p < 4) {
#pragma unroll
      for (int j = 0; j < 8; ++j) us[j] += hr[j];
    } else {
#pragma unroll
      for (int j = 0; j < 8; ++j) is_[j] += hr[j];
    }
  }
  float dot = 0.f;
#pragma unroll
  for (int j = 0; j < 8; ++j) dot += (us[j] * 0.25f) * (is_[j] * (1.f / 6.f));
#pragma unroll
  for (int off = 16; off > 0; off >>= 1) dot += __shfl_down(dot, off, 32);
  if (lane == 0) y[g] = sigm(dot + sw[g]);
}

// ----------------------------------------------------------------- launcher
extern "C" void kernel_launch(void* const* d_in, const int* in_sizes, int n_in,
                              void* d_out, int out_size, void* d_ws, size_t ws_size,
                              hipStream_t stream) {
  const int*   node_id = (const int*)d_in[0];
  const int*   batch   = (const int*)d_in[1];
  const int*   inner   = (const int*)d_in[3];   // [0:NEI] src, [NEI:2NEI] dst
  const int*   outer   = (const int*)d_in[4];   // [0:NEO] src, [NEO:2NEO] dst
  const float* emb     = (const float*)d_in[5];
  const float* nwt     = (const float*)d_in[6];
  const float* lin1_w  = (const float*)d_in[7];
  const float* lin1_b  = (const float*)d_in[8];
  const float* lin2_w  = (const float*)d_in[9];
  const float* lin2_b  = (const float*)d_in[10];
  const float* w_ih    = (const float*)d_in[11];
  const float* w_hh    = (const float*)d_in[12];
  const float* b_ih    = (const float*)d_in[13];
  const float* b_hh    = (const float*)d_in[14];
  const float* h0      = (const float*)d_in[15];
  float* y = (float*)d_out;

  char* ws = (char*)d_ws;
  size_t o = 0;
  auto alloc = [&](size_t bytes) -> void* {
    void* p = ws + o; o += (bytes + 255) & ~(size_t)255; return p;
  };
  float*  x      = (float*)alloc((size_t)NT * GDIM * 4);
  float*  hbuf   = (float*)alloc((size_t)NT * GDIM * 4);
  float*  innerS = (float*)alloc((size_t)NT * GDIM * 4);
  float*  outerS = (float*)alloc((size_t)NT * GDIM * 4);
  float*  sumw   = (float*)alloc((size_t)NB * 4);
  __bf16* w1b    = (__bf16*)alloc((size_t)GHID * GDIM * 2);
  __bf16* w2b    = (__bf16*)alloc((size_t)GDIM * GHID * 2);
  __bf16* wihb   = (__bf16*)alloc((size_t)3 * GDIM * GDIM * 2);
  __bf16* whhb   = (__bf16*)alloc((size_t)3 * GDIM * GDIM * 2);

  long nz = (long)NT * GDIM;
  zero_kernel<<<(int)((nz + 255) / 256), 256, 0, stream>>>(innerS, nz);
  zero_kernel<<<(int)((nz + 255) / 256), 256, 0, stream>>>(outerS, nz);
  zero_kernel<<<(NB + 255) / 256, 256, 0, stream>>>(sumw, NB);

  gather_kernel<<<(NT * 64) / 256, 256, 0, stream>>>((const float4*)emb, node_id, (float4*)x);
  copy4_kernel<<<(NT * 64) / 256, 256, 0, stream>>>((const float4*)h0, (float4*)hbuf, (long)NT * 64);
  nodew_kernel<<<(NT + 255) / 256, 256, 0, stream>>>(node_id, batch, nwt, sumw);

  cvt_kernel<<<(GHID * GDIM + 255) / 256, 256, 0, stream>>>(lin1_w, w1b, GHID * GDIM);
  cvt_kernel<<<(GDIM * GHID + 255) / 256, 256, 0, stream>>>(lin2_w, w2b, GDIM * GHID);
  cvt_kernel<<<(3 * GDIM * GDIM + 255) / 256, 256, 0, stream>>>(w_ih, wihb, 3 * GDIM * GDIM);
  cvt_kernel<<<(3 * GDIM * GDIM + 255) / 256, 256, 0, stream>>>(w_hh, whhb, 3 * GDIM * GDIM);

  // 172032 edges / 16 per wave / 8 waves per block = 1344 blocks (exact)
  edge_mlp_kernel<<<NEI / 16 / 8, 256, 0, stream>>>(x, inner, inner + NEI,
                                                    w1b, lin1_b, w2b, lin2_b, innerS);
  outer_kernel<<<(NEO * 32) / 256, 256, 0, stream>>>(x, outer, outer + NEO, outerS);

  // 40960 nodes / 16 per wave / 8 waves per block = 320 blocks (exact)
  gru_kernel<<<NT / 16 / 8, 256, 0, stream>>>(x,      hbuf, wihb, whhb, b_ih, b_hh, 0);
  gru_kernel<<<NT / 16 / 8, 256, 0, stream>>>(innerS, hbuf, wihb, whhb, b_ih, b_hh, 1);
  gru_kernel<<<NT / 16 / 8, 256, 0, stream>>>(outerS, hbuf, wihb, whhb, b_ih, b_hh, 2);

  final_kernel<<<NB / 8, 256, 0, stream>>>(hbuf, sumw, y);
}
